// GroupedQueryAttention_80436147520130
// MI455X (gfx1250) — matrix operand order repaired and checked by reading, not compiled, not run on hardware
//
#include <hip/hip_runtime.h>

// ---------------------------------------------------------------------------
// MI455X (gfx1250) fused multi-head attention forward (wave32, WMMA bf16).
//   out  = softmax((XWq^T+bq)(XWk^T+bk)^T / sqrt(D)) (XWv^T+bv) Wo^T + bo
// Memory-bound by the 537MB attention-weight output => bf16 WMMA math,
// single write of weights, K/V kept L2-resident (16MB << 192MB L2).
// ---------------------------------------------------------------------------

typedef __bf16 bf16;
typedef __attribute__((ext_vector_type(16))) __bf16 v16bf;
typedef __attribute__((ext_vector_type(8)))  __bf16 v8bf;
typedef __attribute__((ext_vector_type(2)))  __bf16 v2bf;
typedef __attribute__((ext_vector_type(8)))  float  v8f;
typedef __attribute__((ext_vector_type(4)))  float  v4f;

constexpr int kB  = 2;
constexpr int kN  = 2048;
constexpr int kE  = 1024;
constexpr int kH  = 16;
constexpr int kD  = 64;
constexpr int kBN = kB * kN;

#if defined(__has_builtin)
#if __has_builtin(__builtin_amdgcn_cvt_pk_bf16_f32)
#define HAVE_PK_BF16 1
#endif
#endif

__device__ __forceinline__ bf16 f2bf(float f) {
  return static_cast<bf16>(f);           // native bf16 convert path
}

// packed 2x f32 -> bf16 (single v_cvt_pk_bf16_f32 when available)
__device__ __forceinline__ v2bf pk2(float a, float b) {
#ifdef HAVE_PK_BF16
  return __builtin_amdgcn_cvt_pk_bf16_f32(a, b);
#else
  v2bf r; r[0] = static_cast<bf16>(a); r[1] = static_cast<bf16>(b); return r;
#endif
}

__device__ __forceinline__ v16bf mk16(v8bf lo, v8bf hi) {
  v16bf r;
#pragma unroll
  for (int i = 0; i < 8; ++i) { r[i] = lo[i]; r[i + 8] = hi[i]; }
  return r;
}

__device__ __forceinline__ v8f wmma_bf16(v16bf a, v16bf b, v8f c) {
  // emits v_wmma_f32_16x16x32_bf16 (D = A x B + C, f32 accum)
  return __builtin_amdgcn_wmma_f32_16x16x32_bf16(false, a, false, b,
                                                 (short)0, c, false, false);
}

// ---------------------------------------------------------------------------
// GEMM: Y = X @ W^T + bias.  X:[kBN,kE] (f32 or bf16), W:[kE,kE] f32.
// OUT_HND: store bf16 in [B,H,N,D]; else f32 in [kBN,kE].
// Block tile 128x64, 8 waves, each wave 32x32 (2x2 WMMA 16x16 tiles).
// ---------------------------------------------------------------------------
template <bool IN_BF16, bool OUT_HND>
__global__ __launch_bounds__(256) void gemm_kernel(const void* __restrict__ Xv,
                                                   const float* __restrict__ W,
                                                   const float* __restrict__ bias,
                                                   void* __restrict__ Yv) {
  __shared__ alignas(16) bf16 As[128][32];   // A tile, row-major (K contig)
  __shared__ alignas(16) bf16 Bs[64][40];    // B tile, col-major (K contig), padded

  const int tid  = threadIdx.x;
  const int lane = tid & 31;
  const int w    = tid >> 5;        // wave 0..7
  const int wr   = w & 3;           // wave row group (4 x 32 rows)
  const int wc   = w >> 2;          // wave col group (2 x 32 cols)
  const int half = lane >> 4;       // lane half (A/B K-split per ISA layout)
  const int l16  = lane & 15;
  const int row0 = blockIdx.y * 128;
  const int col0 = blockIdx.x * 64;

  const float* Xf = (const float*)Xv;
  const bf16*  Xb = (const bf16*)Xv;

  v8f acc[2][2] = {};

  for (int kk = 0; kk < kE; kk += 32) {
    // ---- stage A tile (convert f32->bf16 on the fly) -----------------------
    {
      const int r    = tid >> 1;
      const int hcol = (tid & 1) * 16;
      const int grow = row0 + r;
      if (IN_BF16) {
        const v8bf* src = (const v8bf*)(Xb + (size_t)grow * kE + kk + hcol);
        *(v8bf*)(&As[r][hcol])     = src[0];
        *(v8bf*)(&As[r][hcol + 8]) = src[1];
      } else {
        const v4f* src = (const v4f*)(Xf + (size_t)grow * kE + kk + hcol);
        v4f xs0 = src[0], xs1 = src[1], xs2 = src[2], xs3 = src[3];
        v2bf* dst = (v2bf*)(&As[r][hcol]);
        dst[0] = pk2(xs0[0], xs0[1]);
        dst[1] = pk2(xs0[2], xs0[3]);
        dst[2] = pk2(xs1[0], xs1[1]);
        dst[3] = pk2(xs1[2], xs1[3]);
        dst[4] = pk2(xs2[0], xs2[1]);
        dst[5] = pk2(xs2[2], xs2[3]);
        dst[6] = pk2(xs3[0], xs3[1]);
        dst[7] = pk2(xs3[2], xs3[3]);
      }
      // prefetch next K-tile of X (global_prefetch_b8 path)
      if (kk + 32 < kE) {
        const char* nxt = IN_BF16
          ? (const char*)(Xb + (size_t)grow * kE + kk + 32)
          : (const char*)(Xf + (size_t)grow * kE + kk + 32);
        __builtin_prefetch(nxt, 0, 1);
      }
    }
    // ---- stage B tile: Bs[c][k] = W[col0+c][kk+k] (W^T col-major) ----------
    {
      const int c = tid >> 2;
      const int q = (tid & 3) * 8;
      const v4f* src = (const v4f*)(W + (size_t)(col0 + c) * kE + kk + q);
      v4f x0 = src[0], x1 = src[1];
      v2bf* dst = (v2bf*)(&Bs[c][q]);
      dst[0] = pk2(x0[0], x0[1]);
      dst[1] = pk2(x0[2], x0[3]);
      dst[2] = pk2(x1[0], x1[1]);
      dst[3] = pk2(x1[2], x1[3]);
    }
    __syncthreads();

    // ---- fragments per ISA 16-bit layouts ----------------------------------
    v16bf afr[2], bfr[2];
#pragma unroll
    for (int mi = 0; mi < 2; ++mi) {
      const bf16* ap = &As[wr * 32 + mi * 16 + l16][0];
      // lanes 0-15: K 0-7 & 16-23 ; lanes 16-31: K 8-15 & 24-31
      afr[mi] = mk16(*(const v8bf*)(ap + 8 * half),
                     *(const v8bf*)(ap + 16 + 8 * half));
    }
#pragma unroll
    for (int ni = 0; ni < 2; ++ni) {
      const bf16* bp = &Bs[wc * 32 + ni * 16 + l16][8 * half];
      bfr[ni] = mk16(*(const v8bf*)(bp), *(const v8bf*)(bp + 16));
    }
#pragma unroll
    for (int mi = 0; mi < 2; ++mi)
#pragma unroll
      for (int ni = 0; ni < 2; ++ni)
        acc[mi][ni] = wmma_bf16(afr[mi], bfr[ni], acc[mi][ni]);
    __syncthreads();
  }

  // ---- epilogue: bias + store ----------------------------------------------
#pragma unroll
  for (int ni = 0; ni < 2; ++ni) {
    const int col = col0 + wc * 32 + ni * 16 + l16;  // C layout: n = lane%16
    const float bval = bias[col];
#pragma unroll
    for (int mi = 0; mi < 2; ++mi) {
#pragma unroll
      for (int r = 0; r < 8; ++r) {
        const int m    = r + 8 * half;               // C layout: m = r + 8*(lane/16)
        const int grow = row0 + wr * 32 + mi * 16 + m;
        const float val = acc[mi][ni][r] + bval;
        if (OUT_HND) {
          const int bi = grow >> 11, n = grow & (kN - 1);
          const int hi = col >> 6,  di = col & (kD - 1);
          ((bf16*)Yv)[(((size_t)(bi * kH + hi) * kN + n) * kD) + di] = f2bf(val);
        } else {
          ((float*)Yv)[(size_t)grow * kE + col] = val;
        }
      }
    }
  }
}

// ---------------------------------------------------------------------------
// Attention: per block one (b,h) and a 16-query strip.
//   S = Q K^T / 8 (full 16x2048 strip in LDS, f32),
//   softmax rows, stream normalized weights to d_out once,
//   O = P V via WMMA (P rescaled from exp-scores while building A-fragments).
// Dynamic LDS ~137 KB (WGP = 320 KB -> 2 blocks resident).
// ---------------------------------------------------------------------------
__global__ __launch_bounds__(128) void attn_kernel(const bf16* __restrict__ Qh,
                                                   const bf16* __restrict__ Kh,
                                                   const bf16* __restrict__ Vh,
                                                   float* __restrict__ Wgt,
                                                   bf16* __restrict__ Ctx) {
  extern __shared__ char smem_raw[];
  constexpr int SSTR = kN + 16;  // padded row stride (floats)
  float* S   = (float*)smem_raw;                               // 16 x SSTR f32
  bf16*  Qs  = (bf16*)(smem_raw + (size_t)16 * SSTR * 4);      // 16 x 64
  bf16*  VT  = (bf16*)((char*)Qs + 16 * 64 * 2);               // 64 x 40 (V^T tile)
  float* red = (float*)((char*)VT + 64 * 40 * 2);              // 16 x 8

  const int tid  = threadIdx.x;
  const int lane = tid & 31;
  const int w    = tid >> 5;      // 4 waves
  const int half = lane >> 4;
  const int l16  = lane & 15;

  const int blk = blockIdx.x;
  const int qt  = blk & 127;          // N/16 strips
  const int bh  = blk >> 7;           // 0..31
  const int q0  = qt * 16;
  const size_t bhBase = (size_t)bh * kN * kD;

  // ---- load Q strip --------------------------------------------------------
  {
    const int r = tid >> 3, sg = (tid & 7) * 8;
    *(v8bf*)(Qs + r * kD + sg) =
        *(const v8bf*)(Qh + bhBase + (size_t)(q0 + r) * kD + sg);
  }
  __syncthreads();

  v16bf aq[2];
#pragma unroll
  for (int kc = 0; kc < 2; ++kc) {
    const bf16* ap = Qs + l16 * kD + kc * 32;
    aq[kc] = mk16(*(const v8bf*)(ap + 16 * half),
                  *(const v8bf*)(ap + 16 * half + 8));
  }

  // ---- phase 1: scores S = Q K^T * scale (wave w owns cols [512w,512w+512))
  const float scale = 0.125f;  // 1/sqrt(64)
  for (int ct = 0; ct < 32; ++ct) {
    const int cb = w * 512 + ct * 16;
    const bf16* kp = Kh + bhBase + (size_t)(cb + l16) * kD;  // key row, contig d
    v16bf bk0 = mk16(*(const v8bf*)(kp + 16 * half),
                     *(const v8bf*)(kp + 16 * half + 8));
    v16bf bk1 = mk16(*(const v8bf*)(kp + 32 + 16 * half),
                     *(const v8bf*)(kp + 32 + 16 * half + 8));
    v8f c = {};
    c = wmma_bf16(aq[0], bk0, c);
    c = wmma_bf16(aq[1], bk1, c);
#pragma unroll
    for (int r = 0; r < 8; ++r)
      S[(r + 8 * half) * SSTR + cb + l16] = c[r] * scale;
  }
  __syncthreads();

  // ---- phase 2: row softmax (8 threads per row), stream weights ------------
  const int srow = tid >> 3;
  const int sseg = tid & 7;
  float* Sr = S + srow * SSTR;
  const int c0 = sseg * 256;

  float mx = -3.0e38f;
  for (int c = c0; c < c0 + 256; c += 4) {
    v4f v = *(const v4f*)(Sr + c);
    mx = fmaxf(mx, fmaxf(fmaxf(v[0], v[1]), fmaxf(v[2], v[3])));
  }
  red[srow * 8 + sseg] = mx;
  __syncthreads();
  if (sseg == 0) {
#pragma unroll
    for (int i = 1; i < 8; ++i) mx = fmaxf(mx, red[srow * 8 + i]);
    red[srow * 8] = mx;
  }
  __syncthreads();
  const float rmax = red[srow * 8];
  __syncthreads();

  float sum = 0.f;
  for (int c = c0; c < c0 + 256; c += 4) {
    v4f v = *(const v4f*)(Sr + c);
#pragma unroll
    for (int i = 0; i < 4; ++i) { v[i] = __expf(v[i] - rmax); sum += v[i]; }
    *(v4f*)(Sr + c) = v;                 // keep un-normalized exp in LDS
  }
  red[srow * 8 + sseg] = sum;
  __syncthreads();
  if (sseg == 0) {
#pragma unroll
    for (int i = 1; i < 8; ++i) sum += red[srow * 8 + i];
    red[srow * 8] = sum;
  }
  __syncthreads();
  const float inv = 1.0f / red[srow * 8];
  __syncthreads();
  if (sseg == 0) red[srow * 8] = inv;    // stash 1/rowsum for phase 3

  float* gw = Wgt + ((size_t)bh * kN + q0 + srow) * kN;
  for (int c = c0; c < c0 + 256; c += 4) {
    v4f v = *(const v4f*)(Sr + c);
    v4f o = {v[0] * inv, v[1] * inv, v[2] * inv, v[3] * inv};
    *(v4f*)(gw + c) = o;                 // single fp32 write of weights
  }
  __syncthreads();

  // ---- phase 3: O = P V (wave w owns output cols d in [16w,16w+16)) --------
  const float invm = red[l16 * 8];       // 1/rowsum for this lane's row
  v8f co = {};
  for (int kk = 0; kk < kN; kk += 32) {
    {  // stage transposed V tile: VT[d][krel]
      const int vr = tid >> 2;
      const int sg = (tid & 3) * 16;
      const bf16* vp = Vh + bhBase + (size_t)(kk + vr) * kD + sg;
      v8bf x0 = *(const v8bf*)(vp);
      v8bf x1 = *(const v8bf*)(vp + 8);
#pragma unroll
      for (int j = 0; j < 8; ++j) {
        VT[(sg + j) * 40 + vr]     = x0[j];
        VT[(sg + 8 + j) * 40 + vr] = x1[j];
      }
    }
    __syncthreads();

    // A fragment: normalized probabilities, packed-converted on the fly
    v16bf ap;
    {
      const float* sp = S + l16 * SSTR + kk;
      v4f l0 = *(const v4f*)(sp + 8 * half);
      v4f l1 = *(const v4f*)(sp + 8 * half + 4);
      v4f h0 = *(const v4f*)(sp + 16 + 8 * half);
      v4f h1 = *(const v4f*)(sp + 16 + 8 * half + 4);
      v2bf p0 = pk2(l0[0] * invm, l0[1] * invm);
      v2bf p1 = pk2(l0[2] * invm, l0[3] * invm);
      v2bf p2 = pk2(l1[0] * invm, l1[1] * invm);
      v2bf p3 = pk2(l1[2] * invm, l1[3] * invm);
      v2bf p4 = pk2(h0[0] * invm, h0[1] * invm);
      v2bf p5 = pk2(h0[2] * invm, h0[3] * invm);
      v2bf p6 = pk2(h1[0] * invm, h1[1] * invm);
      v2bf p7 = pk2(h1[2] * invm, h1[3] * invm);
      ap[0]  = p0[0]; ap[1]  = p0[1]; ap[2]  = p1[0]; ap[3]  = p1[1];
      ap[4]  = p2[0]; ap[5]  = p2[1]; ap[6]  = p3[0]; ap[7]  = p3[1];
      ap[8]  = p4[0]; ap[9]  = p4[1]; ap[10] = p5[0]; ap[11] = p5[1];
      ap[12] = p6[0]; ap[13] = p6[1]; ap[14] = p7[0]; ap[15] = p7[1];
    }
    const bf16* bp = VT + (w * 16 + l16) * 40 + 8 * half;
    v16bf bv = mk16(*(const v8bf*)bp, *(const v8bf*)(bp + 16));
    co = wmma_bf16(ap, bv, co);
    __syncthreads();
  }

  // ---- store context in [B,N,E] bf16 (e = h*64 + d) ------------------------
  {
    const int d  = w * 16 + l16;
    const int bi = bh >> 4, hi = bh & 15;
    const size_t base = (size_t)bi * kN * kE + (size_t)hi * kD + d;
#pragma unroll
    for (int r = 0; r < 8; ++r) {
      const int m = r + 8 * half;
      Ctx[base + (size_t)(q0 + m) * kE] = f2bf(co[r]);
    }
  }
}

// ---------------------------------------------------------------------------
extern "C" void kernel_launch(void* const* d_in, const int* in_sizes, int n_in,
                              void* d_out, int out_size, void* d_ws, size_t ws_size,
                              hipStream_t stream) {
  (void)in_sizes; (void)n_in; (void)out_size; (void)ws_size;

  const float* q  = (const float*)d_in[0];
  const float* k  = (const float*)d_in[1];
  const float* v  = (const float*)d_in[2];
  const float* Wq = (const float*)d_in[3];
  const float* bq = (const float*)d_in[4];
  const float* Wk = (const float*)d_in[5];
  const float* bk = (const float*)d_in[6];
  const float* Wv = (const float*)d_in[7];
  const float* bv = (const float*)d_in[8];
  const float* Wo = (const float*)d_in[9];
  const float* bo = (const float*)d_in[10];

  float* out = (float*)d_out;                        // [B,N,E]
  float* wgt = out + (size_t)kB * kN * kE;           // [B,H,N,N]

  const size_t hndElems = (size_t)kB * kH * kN * kD; // 4,194,304
  bf16* Qh  = (bf16*)d_ws;
  bf16* Kh  = Qh + hndElems;
  bf16* Vh  = Kh + hndElems;
  bf16* Ctx = Vh + hndElems;                         // [B,N,E] bf16

  dim3 gblk(kE / 64, kBN / 128);                     // (16, 32)
  gemm_kernel<false, true><<<gblk, 256, 0, stream>>>(q, Wq, bq, Qh);
  gemm_kernel<false, true><<<gblk, 256, 0, stream>>>(k, Wk, bk, Kh);
  gemm_kernel<false, true><<<gblk, 256, 0, stream>>>(v, Wv, bv, Vh);

  const size_t smem = (size_t)16 * (kN + 16) * 4 + 16 * 64 * 2 + 64 * 40 * 2 +
                      16 * 8 * 4;                    // ~137 KB dynamic LDS
  attn_kernel<<<dim3(kB * kH * (kN / 16)), 128, smem, stream>>>(Qh, Kh, Vh, wgt, Ctx);

  gemm_kernel<true, false><<<gblk, 256, 0, stream>>>(Ctx, Wo, bo, out);
}
